// FLGC_85864986181910
// MI455X (gfx1250) — compile-verified, matrix operand order checked by reading, not run on hardware
//
#include <hip/hip_runtime.h>
#include <hip/hip_bf16.h>
#include <math.h>

typedef __attribute__((ext_vector_type(16))) __bf16        v16bf;
typedef __attribute__((ext_vector_type(8)))  float         v8f;
typedef __attribute__((ext_vector_type(8)))  unsigned int  v8u;
typedef __attribute__((ext_vector_type(4)))  unsigned int  v4u;
typedef __attribute__((ext_vector_type(8)))  int           v8i;
typedef __attribute__((ext_vector_type(4)))  int           v4i;

#define CIN    256
#define COUT   256
#define IMH    56
#define IMW    56
#define HW     3136              // 56*56
#define BATCH  32
#define NTOT   (BATCH*HW)        // 100352
#define KMAX   (9*CIN)           // 2304 : max (and row stride of) packed K
#define NGRP   8

#define BM     32                // rows per group M-tile
#define BN     256               // spatial tile
#define BK     32
#define LDAH   40                // padded halfword stride of LDS tiles (64B row + 16B pad)
#define MAXTILE 16               // >= sum_g ceil(cnt_t[g]/32) (<=15)
#define ROWPAD  32               // zeroed slack rows after row 255 for TDM overreach

static __device__ __forceinline__ unsigned short f32_to_bf16(float f) {
  unsigned int u = __float_as_uint(f);
  u += 0x7FFFu + ((u >> 16) & 1u);       // round to nearest even
  return (unsigned short)(u >> 16);
}

// Load one 16x32 bf16 fragment (A-style layout) from an LDS row pointer.
static __device__ __forceinline__ v16bf ld_frag(const unsigned short* rowp, int hs) {
  const uint4* p0 = (const uint4*)(rowp + hs * 8);
  const uint4* p1 = (const uint4*)(rowp + 16 + hs * 8);
  uint4 a = *p0, b = *p1;
  union { v8u u; v16bf v; } f;
  v8u t = {a.x, a.y, a.z, a.w, b.x, b.y, b.z, b.w};
  f.u = t;
  return f.v;
}

// ---------------- prep: softmax/argmax + all group bookkeeping ---------------
__global__ void flgc_prep(const float* __restrict__ S, const float* __restrict__ T,
                          float* __restrict__ s_scale, float* __restrict__ t_scale,
                          int* __restrict__ perm_arr, int* __restrict__ row_g,
                          int* __restrict__ row_dst, int* __restrict__ inch_pad,
                          int* __restrict__ s_cntg, int* __restrict__ csz_g,
                          int* __restrict__ pofs_g,
                          int* __restrict__ tile_g, int* __restrict__ tile_p0,
                          int* __restrict__ tile_pend, int* __restrict__ ntiles_p) {
  __shared__ int perm_s[COUT];
  __shared__ int tg_s[COUT];
  __shared__ int sg_s[CIN];
  __shared__ int dst_sh[COUT];
  const int c = threadIdx.x;
  { // S row: group + winning softmax prob = 1/sum(exp(v-max))
    float v[NGRP]; float m = -1e30f;
    #pragma unroll
    for (int g = 0; g < NGRP; ++g) { v[g] = S[c * NGRP + g]; m = fmaxf(m, v[g]); }
    float sum = 0.f; int am = 0; float best = -1e30f;
    #pragma unroll
    for (int g = 0; g < NGRP; ++g) {
      sum += __expf(v[g] - m);
      if (v[g] > best) { best = v[g]; am = g; }   // first max, like jnp.argmax
    }
    sg_s[c] = am; s_scale[c] = 1.f / sum;
  }
  { // T row
    float v[NGRP]; float m = -1e30f;
    #pragma unroll
    for (int g = 0; g < NGRP; ++g) { v[g] = T[c * NGRP + g]; m = fmaxf(m, v[g]); }
    float sum = 0.f; int am = 0; float best = -1e30f;
    #pragma unroll
    for (int g = 0; g < NGRP; ++g) {
      sum += __expf(v[g] - m);
      if (v[g] > best) { best = v[g]; am = g; }
    }
    tg_s[c] = am; t_scale[c] = 1.f / sum;
  }
  __syncthreads();
  if (c == 0) {
    // ---- output side: stable group-major perm, scatter map, tile table ----
    int cntT[NGRP]; for (int g = 0; g < NGRP; ++g) cntT[g] = 0;
    for (int o = 0; o < COUT; ++o) cntT[tg_s[o]]++;
    int ofsT[NGRP]; int acc = 0;
    for (int g = 0; g < NGRP; ++g) { ofsT[g] = acc; acc += cntT[g]; }
    int tmp[NGRP]; for (int g = 0; g < NGRP; ++g) tmp[g] = ofsT[g];
    for (int o = 0; o < COUT; ++o) perm_s[tmp[tg_s[o]]++] = o;
    for (int j = 0; j < COUT; ++j) dst_sh[perm_s[perm_s[j]]] = j; // out[:,j]=conv[:,perm[perm[j]]]
    for (int p = 0; p < COUT; ++p) {
      perm_arr[p] = perm_s[p];
      row_g[p]    = tg_s[perm_s[p]];
      row_dst[p]  = dst_sh[perm_s[p]];
    }
    int nt = 0;
    for (int g = 0; g < NGRP; ++g)
      for (int r = 0; r < cntT[g]; r += BM) {
        tile_g[nt] = g; tile_p0[nt] = ofsT[g] + r; tile_pend[nt] = ofsT[g] + cntT[g]; ++nt;
      }
    *ntiles_p = nt;
    // ---- input side: group-sorted channels padded to multiples of 32 ----
    int cntS[NGRP]; for (int g = 0; g < NGRP; ++g) cntS[g] = 0;
    for (int i = 0; i < CIN; ++i) cntS[sg_s[i]]++;
    int cszL[NGRP], pofsL[NGRP]; acc = 0;
    for (int g = 0; g < NGRP; ++g) {
      cszL[g] = ((cntS[g] + 31) / 32) * 32;
      pofsL[g] = acc; acc += cszL[g];
    }
    for (int i = 0; i < acc; ++i) inch_pad[i] = 0;     // dummy channel for pads
    int tmp2[NGRP]; for (int g = 0; g < NGRP; ++g) tmp2[g] = 0;
    for (int i = 0; i < CIN; ++i) {
      int g = sg_s[i];
      inch_pad[pofsL[g] + tmp2[g]++] = i;              // stable
    }
    for (int g = 0; g < NGRP; ++g) {
      s_cntg[g] = cntS[g]; csz_g[g] = cszL[g]; pofs_g[g] = pofsL[g];
    }
  }
}

// ---- pack: rows in perm order, K compacted per group (k = tap*csz + j) -----
__global__ void flgc_pack(const float* __restrict__ conv,
                          const float* __restrict__ s_scale,
                          const float* __restrict__ t_scale,
                          const int* __restrict__ perm_arr,
                          const int* __restrict__ row_g,
                          const int* __restrict__ inch_pad,
                          const int* __restrict__ s_cntg,
                          const int* __restrict__ csz_g,
                          const int* __restrict__ pofs_g,
                          unsigned short* __restrict__ Wg) {
  const int p = blockIdx.x, tid = threadIdx.x;
  if (p >= COUT) {                                   // zeroed slack rows for TDM
    for (int k = tid; k < KMAX; k += blockDim.x) Wg[(long)p * KMAX + k] = 0;
    return;
  }
  const int g = row_g[p];
  const int o = perm_arr[p];
  const int cszg = csz_g[g], cs = s_cntg[g], po = pofs_g[g];
  const float ts = t_scale[o];
  for (int j = tid; j < cszg; j += blockDim.x) {
    const int cc = inch_pad[po + j];
    const float sc = (j < cs) ? ts * s_scale[cc] : 0.f;   // gate implicit: same group
    const float* src = conv + ((long)o * CIN + cc) * 9;
    unsigned short* drow = Wg + (long)p * KMAX + j;
    #pragma unroll
    for (int tap = 0; tap < 9; ++tap)
      drow[tap * cszg] = f32_to_bf16(src[tap] * sc);
  }
}

// ---------------- conv: grouped implicit GEMM, 32x256xK_g tiles -------------
// A tile (32 rows x 64B, row stride KMAX*2) staged by TDM with hardware row
// padding (pad_interval=3: every 64B, pad_amount=3: +16B) -> LDAH=40 layout.
__global__ void __launch_bounds__(256) flgc_conv(
    const float* __restrict__ x, const unsigned short* __restrict__ Wg,
    const int* __restrict__ row_dst, const int* __restrict__ inch_pad,
    const int* __restrict__ csz_g, const int* __restrict__ pofs_g,
    const int* __restrict__ tile_g, const int* __restrict__ tile_p0,
    const int* __restrict__ tile_pend, const int* __restrict__ ntiles_p,
    float* __restrict__ out) {
  __shared__ __align__(16) unsigned short As[2][BM * LDAH];
  __shared__ __align__(16) unsigned short Bs[2][BN * LDAH];
  __shared__ __align__(16) int rd_sh[BM];          // dest channel per tile row

  const int tile = blockIdx.y;
  if (tile >= *ntiles_p) return;                    // surplus worst-case blocks
  const int g    = tile_g[tile];
  const int p0   = tile_p0[tile];
  const int pend = tile_pend[tile];
  const int cszg = csz_g[g];
  const int po   = pofs_g[g];
  const int KCg  = (cszg >> 5) * 9;                 // chunks: 9*csz/32

  const int tid = threadIdx.x;
  const int n0 = blockIdx.x * BN;

  // stage the 32 scatter targets once (invalid pad rows -> -1)
  if (tid < BM) {
    const int p = p0 + tid;
    rd_sh[tid] = (p < pend) ? row_dst[p] : -1;
  }

  // B staging map: 2 group-channels x 16 consecutive n per thread
  const int sc2 = (tid >> 4) * 2;                   // even local k 0..30
  const int sn0 = (tid & 15) * 16;                  // local n base

  int pb[16], ph[16], pw[16];
  #pragma unroll
  for (int i = 0; i < 16; ++i) {
    int ng = n0 + sn0 + i;                          // < NTOT (exact tiling)
    pb[i] = ng / HW;
    int r = ng - pb[i] * HW;
    ph[i] = r / IMW;
    pw[i] = r - ph[i] * IMW;
  }

  // compute map: 8 waves, each owns full 32 M x 32 N (2x2 wmma tiles)
  const int wid = tid >> 5, lane = tid & 31;
  const int lr = lane & 15, hs = lane >> 4;

  v8f acc[2][2];
  #pragma unroll
  for (int ms = 0; ms < 2; ++ms)
    #pragma unroll
    for (int ns = 0; ns < 2; ++ns)
      #pragma unroll
      for (int j = 0; j < 8; ++j) acc[ms][ns][j] = 0.f;

  // ---- invariant TDM descriptor parts, hoisted out of the K loop ----
  const unsigned ldsA[2] = {
    (unsigned)(unsigned long long)(uintptr_t)&As[0][0],
    (unsigned)(unsigned long long)(uintptr_t)&As[1][0]
  };
  v8i g1c;
  g1c[0] = (1 << 16) | (1 << 20) | (3 << 22) | (3 << 25); // 2B elems, pad 64B+16B
  g1c[1] = (int)(32u << 16);           // tensor_dim0 = 32
  g1c[2] = (int)(32u << 16);           // tensor_dim1 = 32
  g1c[3] = (int)(32u << 16);           // tile_dim0 = 32
  g1c[4] = BM;                         // tile_dim1 = 32
  g1c[5] = (int)KMAX;                  // row stride = 2304 elements
  g1c[6] = 0;
  g1c[7] = 0;
  const v4i z4 = {0, 0, 0, 0};
  const v8i z8 = {0, 0, 0, 0, 0, 0, 0, 0};

  unsigned long long gaCur =
      (unsigned long long)(uintptr_t)(Wg + (long)p0 * KMAX);
  int s_tap = 0, s_jj = 0;                          // next chunk to stage

  auto stageA = [&](int buf) {                      // TDM, issued by wave 0
    if (wid == 0) {
      v4u g0;
      g0.x = 1u;                                                // count=1, load
      g0.y = ldsA[buf];
      g0.z = (unsigned)(gaCur & 0xFFFFFFFFull);
      g0.w = (unsigned)((gaCur >> 32) & 0x01FFFFFFull) | (2u << 30); // type=2
      __builtin_amdgcn_tensor_load_to_lds(g0, g1c, z4, z4, z8, 0);
    }
    gaCur += BK * 2;                                // next 64B chunk
  };

  auto stageB = [&](int buf) {                      // x gather via inch_pad
    const int tap = s_tap;
    const int dy = tap / 3 - 1;
    const int dx = tap - (tap / 3) * 3 - 1;
    const int ja = po + s_jj + sc2;
    const long ba = (long)inch_pad[ja] * HW;
    const long bb = (long)inch_pad[ja + 1] * HW;
    #pragma unroll
    for (int i = 0; i < 16; ++i) {
      const int ih = ph[i] + dy, iw = pw[i] + dx;
      float v0 = 0.f, v1 = 0.f;
      if ((unsigned)ih < (unsigned)IMH && (unsigned)iw < (unsigned)IMW) {
        const long base = (long)pb[i] * CIN * HW + ih * IMW + iw;
        v0 = x[base + ba]; v1 = x[base + bb];
      }
      unsigned pk = (unsigned)f32_to_bf16(v0) | ((unsigned)f32_to_bf16(v1) << 16);
      *(unsigned*)&Bs[buf][(sn0 + i) * LDAH + sc2] = pk;
    }
    s_jj += BK;
    if (s_jj >= cszg) { s_jj = 0; s_tap++; }
  };

  if (KCg > 0) {
    stageA(0);
    stageB(0);
    if (wid == 0) __builtin_amdgcn_s_wait_tensorcnt(0);
    __syncthreads();

    for (int kc = 0; kc < KCg; ++kc) {
      const int buf = kc & 1;
      if (kc + 1 < KCg) { stageA(buf ^ 1); stageB(buf ^ 1); }

      v16bf af[2], bfm[2];
      #pragma unroll
      for (int ms = 0; ms < 2; ++ms)
        af[ms] = ld_frag(&As[buf][(ms * 16 + lr) * LDAH], hs);
      #pragma unroll
      for (int ns = 0; ns < 2; ++ns)
        bfm[ns] = ld_frag(&Bs[buf][(wid * 32 + ns * 16 + lr) * LDAH], hs);

      #pragma unroll
      for (int ms = 0; ms < 2; ++ms)
        #pragma unroll
        for (int ns = 0; ns < 2; ++ns)
          acc[ms][ns] = __builtin_amdgcn_wmma_f32_16x16x32_bf16(
              false, af[ms], false, bfm[ns], (short)0, acc[ms][ns], false, false);

      if (wid == 0) __builtin_amdgcn_s_wait_tensorcnt(0);
      __syncthreads();
    }
  }
  __syncthreads();   // rd_sh visible on all paths (incl. KCg==0)

  // store valid rows only, scattered to final output channels
  #pragma unroll
  for (int ms = 0; ms < 2; ++ms) {
    const int4* q = (const int4*)&rd_sh[ms * 16 + hs * 8];
    const int4 d0 = q[0], d1 = q[1];
    const int dsts[8] = {d0.x, d0.y, d0.z, d0.w, d1.x, d1.y, d1.z, d1.w};
    #pragma unroll
    for (int ns = 0; ns < 2; ++ns) {
      const int ng = n0 + wid * 32 + ns * 16 + lr;
      const int b = ng / HW;
      const int r = ng - b * HW;
      float* pob = out + (long)b * COUT * HW + r;
      #pragma unroll
      for (int j = 0; j < 8; ++j)
        if (dsts[j] >= 0) pob[(long)dsts[j] * HW] = acc[ms][ns][j];
    }
  }
}

extern "C" void kernel_launch(void* const* d_in, const int* in_sizes, int n_in,
                              void* d_out, int out_size, void* d_ws, size_t ws_size,
                              hipStream_t stream) {
  const float* x    = (const float*)d_in[0];
  const float* conv = (const float*)d_in[1];
  const float* S    = (const float*)d_in[2];
  const float* T    = (const float*)d_in[3];
  float* out = (float*)d_out;

  char* ws = (char*)d_ws;
  unsigned short* Wg = (unsigned short*)ws;   // (256+32) x 2304 bf16
  size_t off = (size_t)(COUT + ROWPAD) * KMAX * sizeof(unsigned short);
  float* s_scale  = (float*)(ws + off); off += COUT * sizeof(float);
  float* t_scale  = (float*)(ws + off); off += COUT * sizeof(float);
  int* perm_arr   = (int*)(ws + off);   off += COUT * sizeof(int);
  int* row_g      = (int*)(ws + off);   off += COUT * sizeof(int);
  int* row_dst    = (int*)(ws + off);   off += COUT * sizeof(int);
  int* inch_pad   = (int*)(ws + off);   off += 512  * sizeof(int);
  int* s_cntg     = (int*)(ws + off);   off += NGRP * sizeof(int);
  int* csz_g      = (int*)(ws + off);   off += NGRP * sizeof(int);
  int* pofs_g     = (int*)(ws + off);   off += NGRP * sizeof(int);
  int* tile_g     = (int*)(ws + off);   off += MAXTILE * sizeof(int);
  int* tile_p0    = (int*)(ws + off);   off += MAXTILE * sizeof(int);
  int* tile_pend  = (int*)(ws + off);   off += MAXTILE * sizeof(int);
  int* ntiles_p   = (int*)(ws + off);   off += sizeof(int);

  flgc_prep<<<1, 256, 0, stream>>>(S, T, s_scale, t_scale, perm_arr, row_g,
                                   row_dst, inch_pad, s_cntg, csz_g, pofs_g,
                                   tile_g, tile_p0, tile_pend, ntiles_p);
  flgc_pack<<<COUT + ROWPAD, 256, 0, stream>>>(conv, s_scale, t_scale, perm_arr,
                                               row_g, inch_pad, s_cntg, csz_g,
                                               pofs_g, Wg);
  dim3 grid(NTOT / BN, MAXTILE);
  flgc_conv<<<grid, 256, 0, stream>>>(x, Wg, row_dst, inch_pad, csz_g, pofs_g,
                                      tile_g, tile_p0, tile_pend, ntiles_p, out);
}
